// RelationalFeatureExtractor_71339406786593
// MI455X (gfx1250) — matrix operand order repaired and checked by reading, not compiled, not run on hardware
//
#include <hip/hip_runtime.h>

// ---------------------------------------------------------------------------
// RelationalFeatureExtractor for MI455X (gfx1250, wave32, WMMA)
//   out(B=8, H=8, Nq=256, Nk=256) f32
//   GEMM1: M=pairs, K=64 feats, N=256 hidden  (v_wmma_f32_16x16x32_f16)
//   GEMM2: M=pairs, K=256 hidden, N=16 (8 heads + 8 zero pad)
// Fast-math VALU: v_rcp_f32 / v_sqrt_f32 / v_exp_f32 instead of IEEE div/libm.
// ---------------------------------------------------------------------------

typedef __attribute__((ext_vector_type(16))) _Float16 v16h;
typedef __attribute__((ext_vector_type(8)))  float    v8f;

#define NQ   256
#define NKK  256
#define SDIM 10
#define HID  256
#define NH   8

// LDS row strides (in halfs / floats), padded so 16-lane row fans hit
// distinct banks, and kept multiples of 8 halfs (16B) for b128 alignment.
#define SA_STRIDE  72    // features:  128 rows x 64 halfs
#define ST_STRIDE  40    // stage:     16 rows x 32 halfs per wave
#define SW2_STRIDE 264   // W2t:       16 rows x 256 halfs
#define SO_STRIDE  132   // out stage: 8 heads x 128 floats

__device__ __forceinline__ float fast_rcp(float x)  { return __builtin_amdgcn_rcpf(x); }
__device__ __forceinline__ float fast_sqrt(float x) { return __builtin_amdgcn_sqrtf(x); }
// tanh for r >= 0: 1 - 2/(e^{2r} + 1); one v_exp + one v_rcp
__device__ __forceinline__ float fast_tanh_pos(float r) {
  return 1.0f - 2.0f * fast_rcp(__expf(r + r) + 1.0f);
}
__device__ __forceinline__ float fast_silu(float x) {
  return x * fast_rcp(1.0f + __expf(-x));
}

__device__ __forceinline__ void fourier4(float v, float* f) {
  float p = v;
#pragma unroll
  for (int i = 0; i < 4; ++i) { f[i] = __sinf(p); f[4 + i] = __cosf(p); p = p + p; }
}
__device__ __forceinline__ void fourier2(float v, float* f) {
  float p = v;
#pragma unroll
  for (int i = 0; i < 2; ++i) { f[i] = __sinf(p); f[2 + i] = __cosf(p); p = p + p; }
}

// Prep: W1 (64x256) -> W1t f16 [n=256][k=64]; W2 (256x8) -> W2t f16 [n=16 pad][k=256]
__global__ void rfe_prep_weights(const float* __restrict__ W1,
                                 const float* __restrict__ W2,
                                 _Float16* __restrict__ w1t,
                                 _Float16* __restrict__ w2t) {
  int t = blockIdx.x * 256 + threadIdx.x;
  if (t < 64 * 256) {
    int n = t >> 6, k = t & 63;
    w1t[t] = (_Float16)W1[k * 256 + n];
  }
  int u = t - 64 * 256;
  if (u >= 0 && u < 16 * 256) {
    int n = u >> 8, k = u & 255;
    w2t[u] = (_Float16)((n < NH) ? W2[k * NH + n] : 0.0f);
  }
}

__global__ __launch_bounds__(256)
void rfe_main_kernel(const float* __restrict__ Q,
                     const float* __restrict__ Kst,
                     const _Float16* __restrict__ w1t,
                     const _Float16* __restrict__ w2t,
                     const float* __restrict__ b1g,
                     const float* __restrict__ b2g,
                     float* __restrict__ out) {
  __shared__ __align__(16) _Float16 sA[128 * SA_STRIDE];        // 18432 B
  __shared__ __align__(16) _Float16 sStage[8 * 16 * ST_STRIDE]; // 10240 B
  __shared__ __align__(16) _Float16 sW2[16 * SW2_STRIDE];       //  8448 B
  __shared__ __align__(16) float    sOut[NH * SO_STRIDE];       //  4224 B

  const int tid   = threadIdx.x;
  const int khalf = blockIdx.x;   // 0..1  (128 k's each)
  const int q     = blockIdx.y;   // 0..255
  const int b     = blockIdx.z;   // 0..7

  // Warm GL2 with the f16 weights (global_prefetch_b8).
  __builtin_prefetch(w1t + tid * 64, 0, 0);

  // ---- Stage W2t (f16, padded to 16 heads) into LDS: 512 uint4 ----
  {
    const uint4* src = (const uint4*)w2t;
#pragma unroll
    for (int i = 0; i < 2; ++i) {
      int idx = tid + i * 256;          // 0..511
      int row = idx >> 5, c = idx & 31; // 32 uint4 per 256-half row
      *(uint4*)&sW2[row * SW2_STRIDE + c * 8] = src[row * 32 + c];
    }
  }

  // ---- Feature computation: threads 0..127, one (q,k) pair each ----
  const float* qs = Q + ((size_t)b * NQ + q) * SDIM;  // uniform -> scalar loads
  const float qtm = qs[0], qpx = qs[3], qpy = qs[4];
  const float qvx = qs[5], qvy = qs[6], qax = qs[7], qay = qs[8];

  if (tid < 128) {
    const int m = tid;
    const float* ks = Kst + ((size_t)b * NKK + khalf * 128 + m) * SDIM;
    const float ktm = ks[0], kpx = ks[3], kpy = ks[4];
    const float kvx = ks[5], kvy = ks[6], kax = ks[7], kay = ks[8];

    const float dpx = kpx - qpx, dpy = kpy - qpy;
    const float dvx = kvx - qvx, dvy = kvy - qvy;
    const float dist     = fast_sqrt(dpx * dpx + dpy * dpy + 1e-6f);
    const float inv_dist = fast_rcp(dist + 0.1f);
    const float ib       = fast_rcp(dist + 1e-6f);
    const float bx = dpx * ib, by = dpy * ib;
    const float ata    = bx * qax + by * qay;
    const float aspect = bx * kax + by * kay;
    const float dotdpv = dpx * dvx + dpy * dvy;
    const float spd2   = dvx * dvx + dvy * dvy;
    const float ttca   = fast_tanh_pos(fmaxf(0.0f, -dotdpv * fast_rcp(spd2 + 1e-6f)));
    const float same   = (qtm == ktm) ? 1.0f : 0.0f;
    const float dspd   = fast_sqrt(kvx * kvx + kvy * kvy)
                       - fast_sqrt(qvx * qvx + qvy * qvy);

    float f[64];
    fourier4(dist,     f + 0);  fourier4(inv_dist, f + 8);
    fourier4(ata,      f + 16); fourier4(aspect,   f + 24);
    fourier2(dpx,      f + 32); fourier2(dpy,      f + 36);
    fourier2(dvx,      f + 40); fourier2(dvy,      f + 44);
    fourier2(ttca,     f + 48); fourier2(dist,     f + 52);
    fourier2(same,     f + 56); fourier2(dspd,     f + 60);

    _Float16* dst = &sA[m * SA_STRIDE];
#pragma unroll
    for (int c = 0; c < 8; ++c) {   // 8 x ds_store_b128
      union { _Float16 h[8]; uint4 u; } t;
#pragma unroll
      for (int j = 0; j < 8; ++j) t.h[j] = (_Float16)f[c * 8 + j];
      *(uint4*)(dst + c * 8) = t.u;
    }
  }
  __syncthreads();

  // ---- Fused GEMM1 (64->256, SiLU) + GEMM2 (256->16) per wave ----
  const int wv   = tid >> 5;        // wave id: M-tile (16 pairs)
  const int lane = tid & 31;
  const int hsel = lane >> 4;       // lane half-select
  const int lm   = lane & 15;

  const _Float16* aBase = &sA[(wv * 16 + lm) * SA_STRIDE];
  _Float16* stBase      = &sStage[wv * 16 * ST_STRIDE];

  v8f acc2 = (v8f)0.0f;

  for (int kb2 = 0; kb2 < 8; ++kb2) {      // hidden K32 chunks
#pragma unroll
    for (int t16 = 0; t16 < 2; ++t16) {    // two 16-wide hidden n-tiles
      const int n_tile = kb2 * 2 + t16;
      v8f acc = (v8f)0.0f;
#pragma unroll
      for (int kb = 0; kb < 2; ++kb) {     // feature K: 0..31, 32..63
        const int base = kb * 32 + hsel * 8;
        v16h a;
        ((uint4*)&a)[0] = *(const uint4*)(aBase + base);        // K base..+7
        ((uint4*)&a)[1] = *(const uint4*)(aBase + base + 16);   // K base+16..+23
        const _Float16* wp = w1t + (n_tile * 16 + lm) * 64 + kb * 32 + hsel * 8;
        v16h bm;
        ((uint4*)&bm)[0] = ((const uint4*)wp)[0];
        ((uint4*)&bm)[1] = ((const uint4*)wp)[2];
        acc = __builtin_amdgcn_wmma_f32_16x16x32_f16(
            false, a, false, bm, (short)0, acc, false, false);
      }
      const float bv = b1g[n_tile * 16 + lm];  // N = lm for every D vgpr
#pragma unroll
      for (int r = 0; r < 8; ++r) {            // bias + SiLU, stage as A-layout
        const float s = fast_silu(acc[r] + bv);
        stBase[(r + 8 * hsel) * ST_STRIDE + t16 * 16 + lm] = (_Float16)s;
      }
    }
    asm volatile("s_wait_dscnt 0" ::: "memory");  // LDS RAW within wave

    const _Float16* sb = stBase + lm * ST_STRIDE + hsel * 8;
    v16h a2;
    ((uint4*)&a2)[0] = *(const uint4*)(sb);
    ((uint4*)&a2)[1] = *(const uint4*)(sb + 16);
    const _Float16* w2p = &sW2[lm * SW2_STRIDE + kb2 * 32 + hsel * 8];
    v16h b2m;
    ((uint4*)&b2m)[0] = ((const uint4*)w2p)[0];
    ((uint4*)&b2m)[1] = ((const uint4*)w2p)[2];
    acc2 = __builtin_amdgcn_wmma_f32_16x16x32_f16(
        false, a2, false, b2m, (short)0, acc2, false, false);
  }

  // ---- acc2: head = lm (valid < 8), M = r + 8*hsel -> LDS transpose ----
  if (lm < NH) {
    const float b2v = b2g[lm];
#pragma unroll
    for (int r = 0; r < 8; ++r)
      sOut[lm * SO_STRIDE + wv * 16 + r + 8 * hsel] = acc2[r] + b2v;
  }
  __syncthreads();

  // ---- Coalesced writeout: 8 heads x 128 consecutive k each ----
#pragma unroll
  for (int i = 0; i < 4; ++i) {
    const int j = tid + i * 256;       // 0..1023
    const int head = j >> 7, kk = j & 127;
    out[(((size_t)b * NH + head) * NQ + q) * NKK + khalf * 128 + kk] =
        sOut[head * SO_STRIDE + kk];
  }
}

extern "C" void kernel_launch(void* const* d_in, const int* in_sizes, int n_in,
                              void* d_out, int out_size, void* d_ws, size_t ws_size,
                              hipStream_t stream) {
  const float* Q   = (const float*)d_in[0];
  const float* Kst = (const float*)d_in[1];
  const float* W1  = (const float*)d_in[2];
  const float* b1  = (const float*)d_in[3];
  const float* W2  = (const float*)d_in[4];
  const float* b2  = (const float*)d_in[5];
  float* out = (float*)d_out;

  _Float16* w1t = (_Float16*)d_ws;            // 16384 halfs
  _Float16* w2t = w1t + 64 * 256;             //  4096 halfs (40 KB total ws)

  rfe_prep_weights<<<80, 256, 0, stream>>>(W1, W2, w1t, w2t);

  dim3 grid(2, NQ, 8);                        // (khalf, q, b) = 4096 blocks
  rfe_main_kernel<<<grid, 256, 0, stream>>>(Q, Kst, w1t, w2t, b1, b2, out);
}
